// LayerAttentionAggregator_77421080477931
// MI455X (gfx1250) — matrix operand order repaired and checked by reading, not compiled, not run on hardware
//
#include <hip/hip_runtime.h>
#include <hip/hip_bf16.h>

typedef __bf16 bf16;
typedef __attribute__((ext_vector_type(16))) __bf16 v16bf;
typedef __attribute__((ext_vector_type(8)))  __bf16 bf16x8;
typedef __attribute__((ext_vector_type(8)))  float  v8f;
typedef __attribute__((ext_vector_type(4)))  float  f32x4;

#define LDS_STRIDE 40  // bf16 elems: 80B row pitch = 5*16B (keeps b128 alignment, spreads banks)

// ---- staging types: global -> regs (fetch), regs -> LDS bf16 (stash) ----
struct StF { f32x4 lo, hi; };     // f32 source, converted at stash time
struct StB { bf16x8 v;     };     // bf16 source, passthrough

__device__ __forceinline__ StF fetchS(const float* p) {
  StF s; s.lo = *(const f32x4*)p; s.hi = *(const f32x4*)(p + 4); return s;
}
__device__ __forceinline__ StB fetchS(const bf16* p) {
  StB s; s.v = *(const bf16x8*)p; return s;
}
__device__ __forceinline__ bf16x8 toBf16(const StF& s) {
  bf16x8 r;
  r[0]=(bf16)s.lo[0]; r[1]=(bf16)s.lo[1]; r[2]=(bf16)s.lo[2]; r[3]=(bf16)s.lo[3];
  r[4]=(bf16)s.hi[0]; r[5]=(bf16)s.hi[1]; r[6]=(bf16)s.hi[2]; r[7]=(bf16)s.hi[3];
  return r;
}
__device__ __forceinline__ bf16x8 toBf16(const StB& s) { return s.v; }

// =====================================================================================
// Generic batched GEMM: C[z][m,n] = sum_k A[z][m,k] * W[z][n,k]  + bias[z][n]
// A: (M,K) row-major (f32 or bf16), W: (N,K) row-major bf16, C: f32 with row stride ldc.
// BK = 32, bf16 WMMA 16x16x32, f32 accumulate. Double-buffered LDS, 2-deep pipeline,
// one barrier per K-step, exact-trip-count cooperative loads (no exec-mask branching).
// =====================================================================================
template<int BM, int BN, int WGM, int WGN, typename AT>
__global__ __launch_bounds__(WGM*WGN*32)
void gemm_bf16_wmma(const AT* __restrict__ A, long long aBatch,
                    const bf16* __restrict__ W, long long wBatch,
                    const float* __restrict__ bias, long long biasBatch,
                    float* __restrict__ C, long long cBatch, int ldc,
                    int K)
{
  constexpr int NT   = WGM*WGN*32;
  constexpr int WTM  = BM/(WGM*16);
  constexpr int WTN  = BN/(WGN*16);
  constexpr int AIT  = (BM*4)/NT;       // 8-elem chunks per thread for A tile
  constexpr int BIT  = (BN*4)/NT;       // 8-elem chunks per thread for W tile
  constexpr int ABUF = BM*LDS_STRIDE;
  constexpr int BBUF = BN*LDS_STRIDE;

  __shared__ bf16 sA[2*ABUF];
  __shared__ bf16 sB[2*BBUF];

  const int tid  = threadIdx.x;
  const int wave = tid >> 5;
  const int lane = tid & 31;
  const int half = lane >> 4;   // K-half selector per ISA 16-bit layouts
  const int l16  = lane & 15;
  const int waveM = wave / WGN;
  const int waveN = wave % WGN;

  const int z = blockIdx.z;
  A    += (long long)z * aBatch;
  W    += (long long)z * wBatch;
  bias += (long long)z * biasBatch;
  C    += (long long)z * cBatch;

  const long long rowA0 = (long long)blockIdx.x * BM;
  const long long rowW0 = (long long)blockIdx.y * BN;

  using ASt = decltype(fetchS((const AT*)nullptr));
  ASt aReg[AIT];
  StB wReg[BIT];

  auto fetchTile = [&](int k0) {
#pragma unroll
    for (int it = 0; it < AIT; ++it) {
      const int c = tid + it*NT;
      aReg[it] = fetchS(A + (rowA0 + (c >> 2))*(long long)K + k0 + ((c & 3) << 3));
    }
#pragma unroll
    for (int it = 0; it < BIT; ++it) {
      const int c = tid + it*NT;
      wReg[it] = fetchS(W + (rowW0 + (c >> 2))*(long long)K + k0 + ((c & 3) << 3));
    }
  };
  auto stashTile = [&](int buf) {
#pragma unroll
    for (int it = 0; it < AIT; ++it) {
      const int c = tid + it*NT;
      *(bf16x8*)(sA + buf*ABUF + (c >> 2)*LDS_STRIDE + ((c & 3) << 3)) = toBf16(aReg[it]);
    }
#pragma unroll
    for (int it = 0; it < BIT; ++it) {
      const int c = tid + it*NT;
      *(bf16x8*)(sB + buf*BBUF + (c >> 2)*LDS_STRIDE + ((c & 3) << 3)) = toBf16(wReg[it]);
    }
  };

  v8f acc[WTM][WTN];
#pragma unroll
  for (int i = 0; i < WTM; ++i)
#pragma unroll
    for (int j = 0; j < WTN; ++j) acc[i][j] = {};

  const int kTiles = K >> 5;

  // ---- pipeline prologue ----
  fetchTile(0);
  stashTile(0);
  if (kTiles > 1) fetchTile(32);
  __syncthreads();

  for (int kt = 0; kt < kTiles; ++kt) {
    const int cur = kt & 1;
    const int nxt = cur ^ 1;

    // stash tile kt+1 (fetched last iteration), then start fetching tile kt+2
    if (kt + 1 < kTiles) stashTile(nxt);
    if (kt + 2 < kTiles) fetchTile((kt + 2) << 5);

    // ---- compute on buffer `cur` ----
    const bf16* sAc = sA + cur*ABUF;
    const bf16* sBc = sB + cur*BBUF;

    v16bf afrag[WTM];
#pragma unroll
    for (int i = 0; i < WTM; ++i) {
      const bf16* base = sAc + (waveM*WTM*16 + i*16 + l16)*LDS_STRIDE + half*8;
      v16bf a;
      *((bf16x8*)&a)     = *(const bf16x8*)(base);
      *((bf16x8*)&a + 1) = *(const bf16x8*)(base + 16);
      afrag[i] = a;
    }
#pragma unroll
    for (int j = 0; j < WTN; ++j) {
      const bf16* base = sBc + (waveN*WTN*16 + j*16 + l16)*LDS_STRIDE + half*8;
      v16bf b;
      *((bf16x8*)&b)     = *(const bf16x8*)(base);
      *((bf16x8*)&b + 1) = *(const bf16x8*)(base + 16);
#pragma unroll
      for (int i = 0; i < WTM; ++i)
        acc[i][j] = __builtin_amdgcn_wmma_f32_16x16x32_bf16(
            false, afrag[i], false, b, (short)0, acc[i][j], false, false);
    }
    __syncthreads();
  }

  // ---- epilogue: C/D layout -> VGPR p maps to M = p + half*8 ----
#pragma unroll
  for (int i = 0; i < WTM; ++i) {
#pragma unroll
    for (int j = 0; j < WTN; ++j) {
      const long long n     = rowW0 + waveN*WTN*16 + j*16 + l16;
      const long long mBase = rowA0 + waveM*WTM*16 + i*16 + half*8;
      const float bv = bias[n];
#pragma unroll
      for (int p = 0; p < 8; ++p)
        C[(mBase + p)*(long long)ldc + n] = acc[i][j][p] + bv;
    }
  }
}

// =====================================================================================
// LayerNorm over width-768 rows, wave32 per row, f32 in -> bf16 out
// =====================================================================================
template<int WIDTH>
__global__ __launch_bounds__(256)
void layernorm_to_bf16(const float* __restrict__ x, const float* __restrict__ g,
                       const float* __restrict__ bta, bf16* __restrict__ y, int rows)
{
  constexpr int PER = WIDTH / 32;
  const int wave = threadIdx.x >> 5, lane = threadIdx.x & 31;
  const long long row = (long long)blockIdx.x * 8 + wave;
  if (row >= rows) return;
  const float* xr = x + row * WIDTH;
  float v[PER];
  float s = 0.f;
#pragma unroll
  for (int i = 0; i < PER; ++i) { v[i] = xr[lane + i*32]; s += v[i]; }
#pragma unroll
  for (int m = 16; m >= 1; m >>= 1) s += __shfl_xor(s, m, 32);
  const float mean = s * (1.0f / WIDTH);
  float q = 0.f;
#pragma unroll
  for (int i = 0; i < PER; ++i) { float d = v[i] - mean; q += d*d; }
#pragma unroll
  for (int m = 16; m >= 1; m >>= 1) q += __shfl_xor(q, m, 32);
  const float rstd = rsqrtf(q * (1.0f / WIDTH) + 1e-5f);
  bf16* yr = y + row * WIDTH;
#pragma unroll
  for (int i = 0; i < PER; ++i) {
    const int c = lane + i*32;
    yr[c] = (bf16)((v[i] - mean) * rstd * g[c] + bta[c]);
  }
}

// =====================================================================================
// f32 -> bf16 grid-stride convert (weights)
// =====================================================================================
__global__ void cvt_f32_bf16(const float* __restrict__ x, bf16* __restrict__ y, long long n)
{
  long long i = (long long)blockIdx.x * blockDim.x + threadIdx.x;
  const long long stride = (long long)gridDim.x * blockDim.x;
  for (; i < n; i += stride) y[i] = (bf16)x[i];
}

// =====================================================================================
// Per-token attention: scores (H=4 heads, L=13 layers, head_dim=8), softmax over L,
// out_heads[h, 0..767] = sum_l attn[h][l] * KV[l, :]   (KV in bf16)
// =====================================================================================
__global__ __launch_bounds__(256)
void attn_kernel(const float* __restrict__ q, const float* __restrict__ k,
                 const bf16* __restrict__ kv, bf16* __restrict__ outh)
{
  constexpr int L = 13, H = 4, HD = 8, QK = 32, V = 768;
  const int bt = blockIdx.x;
  const int t  = threadIdx.x;
  __shared__ float qs[QK];
  __shared__ float ks[L * QK];
  __shared__ float sAttn[H * L];

  if (t < QK) qs[t] = q[(long long)bt * QK + t];
  for (int i = t; i < L * QK; i += 256) ks[i] = k[(long long)bt * L * QK + i];
  __syncthreads();

  if (t < H * L) {
    const int h = t / L, l = t % L;
    float s = 0.f;
#pragma unroll
    for (int d = 0; d < HD; ++d) s += qs[h*HD + d] * ks[l*QK + h*HD + d];
    s *= 0.17677669529663687f;            // 1/(sqrt(8)*TEMPERATURE)
    if (s != s) s = -INFINITY;            // NaN -> -inf per reference
    sAttn[t] = s;
  }
  __syncthreads();

  if (t < H) {
    float mx = -INFINITY;
    for (int l = 0; l < L; ++l) mx = fmaxf(mx, sAttn[t*L + l]);
    float sum = 0.f;
    for (int l = 0; l < L; ++l) { float e = __expf(sAttn[t*L + l] - mx); sAttn[t*L + l] = e; sum += e; }
    const float inv = 1.f / sum;
    for (int l = 0; l < L; ++l) sAttn[t*L + l] *= inv;
  }
  __syncthreads();

  const bf16* kvr = kv + (long long)bt * L * V;
#pragma unroll
  for (int o = 0; o < 12; ++o) {          // H*V = 3072 = 12 * 256
    const int hv = t + o * 256;
    const int h = hv / V, vc = hv - h * V;
    float acc = 0.f;
#pragma unroll
    for (int l = 0; l < L; ++l) acc += sAttn[h*L + l] * (float)kvr[l*V + vc];
    outh[(long long)bt * (H*V) + hv] = (bf16)acc;
  }
}

extern "C" void kernel_launch(void* const* d_in, const int* in_sizes, int n_in,
                              void* d_out, int out_size, void* d_ws, size_t ws_size,
                              hipStream_t stream)
{
  (void)in_sizes; (void)n_in; (void)out_size; (void)ws_size;
  const float* base  = (const float*)d_in[0];
  const float* reps  = (const float*)d_in[1];
  const float* alW   = (const float*)d_in[2];
  const float* alB   = (const float*)d_in[3];
  const float* qlnG  = (const float*)d_in[4];
  const float* qlnB  = (const float*)d_in[5];
  const float* kvlnG = (const float*)d_in[6];
  const float* kvlnB = (const float*)d_in[7];
  const float* qW    = (const float*)d_in[8];
  const float* qB    = (const float*)d_in[9];
  const float* kW    = (const float*)d_in[10];
  const float* kB    = (const float*)d_in[11];
  const float* outW  = (const float*)d_in[12];
  const float* outB  = (const float*)d_in[13];
  float* out = (float*)d_out;

  constexpr int Bsz = 8, T = 1024, D = 768, R = 768, V = 768, L = 13, QK = 32, H = 4;
  constexpr long long MT  = (long long)Bsz * T;   // 8192 tokens
  constexpr long long MKV = MT * L;               // 106496 (token, layer) rows

  size_t off = 0;
  auto alloc = [&](size_t bytes) -> char* {
    char* p = (char*)d_ws + off;
    off += (bytes + 255) & ~(size_t)255;
    return p;
  };
  float* aligned  = (float*)alloc((size_t)MKV * V * 4);        // (B,T,L,V) pre-LN
  bf16*  kvB      = (bf16*) alloc((size_t)MKV * V * 2);        // LN'd KV, bf16
  bf16*  qlnOut   = (bf16*) alloc((size_t)MT * D * 2);         // LN'd Q, bf16
  bf16*  alWb     = (bf16*) alloc((size_t)L * V * R * 2);
  bf16*  qWb      = (bf16*) alloc((size_t)QK * D * 2);
  bf16*  kWb      = (bf16*) alloc((size_t)QK * V * 2);
  bf16*  outWb    = (bf16*) alloc((size_t)D * H * V * 2);
  float* qproj    = (float*)alloc((size_t)MT * QK * 4);
  float* kproj    = (float*)alloc((size_t)MKV * QK * 4);
  bf16*  outHeads = (bf16*) alloc((size_t)MT * H * V * 2);

  // 1) weights -> bf16 (once)
  cvt_f32_bf16<<<dim3(1024), 256, 0, stream>>>(alW,  alWb,  (long long)L * V * R);
  cvt_f32_bf16<<<dim3(64),   256, 0, stream>>>(qW,   qWb,   (long long)QK * D);
  cvt_f32_bf16<<<dim3(64),   256, 0, stream>>>(kW,   kWb,   (long long)QK * V);
  cvt_f32_bf16<<<dim3(512),  256, 0, stream>>>(outW, outWb, (long long)D * H * V);

  // 2) aligner: aligned[bt, l, v] = reps[l, bt, :] . alW[l][v, :] + alB[l][v]   (batched over l)
  gemm_bf16_wmma<128,128,4,2,float><<<dim3((int)(MT/128), V/128, L), 256, 0, stream>>>(
      reps, (long long)MT * R,
      alWb, (long long)V * R,
      alB,  (long long)V,
      aligned, (long long)V, L * V,
      R);

  // 3) LayerNorms -> bf16
  layernorm_to_bf16<768><<<dim3((int)(MKV/8)), 256, 0, stream>>>(aligned, kvlnG, kvlnB, kvB, (int)MKV);
  layernorm_to_bf16<768><<<dim3((int)(MT/8)),  256, 0, stream>>>(base,    qlnG,  qlnB,  qlnOut, (int)MT);

  // 4) q / k projections (narrow N = 32)
  gemm_bf16_wmma<128,32,4,1,bf16><<<dim3((int)(MT/128), 1, 1), 128, 0, stream>>>(
      qlnOut, 0, qWb, 0, qB, 0, qproj, 0, QK, D);
  gemm_bf16_wmma<128,32,4,1,bf16><<<dim3((int)(MKV/128), 1, 1), 128, 0, stream>>>(
      kvB, 0, kWb, 0, kB, 0, kproj, 0, QK, V);

  // 5) per-token scores + softmax + weighted value sum
  attn_kernel<<<dim3((int)MT), 256, 0, stream>>>(qproj, kproj, kvB, outHeads);

  // 6) out projection -> d_out (f32)
  gemm_bf16_wmma<128,128,4,2,bf16><<<dim3((int)(MT/128), D/128, 1), 256, 0, stream>>>(
      outHeads, 0, outWb, 0, outB, 0, out, 0, D, H * V);
}
